// MoEGate_4939212391142
// MI455X (gfx1250) — compile-verified, matrix operand order checked by reading, not run on hardware
//
#include <hip/hip_runtime.h>
#include <hip/hip_bf16.h>

// MoE gate for MI455X (gfx1250, wave32, WMMA).
// Fused: LayerNorm -> [x @ W1 + b1] -> GELU(exact) -> [h @ W2 + b2]
//        -> softmax(8) -> top-2 -> mask+renorm.
// GEMMs: v_wmma_f32_16x16x32_bf16 (f32 accumulate); epilogues in f32.
// W1 panel staging uses GLOBAL_LOAD_ASYNC_TO_LDS_B128 (ASYNCcnt-tracked,
// no VGPR round-trip) when the toolchain exposes the builtin.

#define D_MODEL   1024
#define N_EXPERTS 8
#define N_TOKENS  262144
#define LN_EPS    1e-5f
#define RENORM_EPS 1e-8f

#define M_TILE 32            // tokens per workgroup (2 WMMA m-tiles)
#define NCHUNK 256           // W1 columns per chunk (16 n-tiles, 4 per wave)
#define KSTEP  32            // WMMA K per step
#define XS_STRIDE  (D_MODEL + 8)   // bf16 elems; +8 pad -> conflict-free ds_b128
#define W1S_STRIDE (KSTEP + 8)     // 40
#define HS_STRIDE  (NCHUNK + 8)    // 264

typedef __attribute__((ext_vector_type(16))) __bf16 v16bf;
typedef __attribute__((ext_vector_type(8)))  float  v8f;
typedef __attribute__((ext_vector_type(4)))  int    v4i;

#if defined(__has_builtin)
#if __has_builtin(__builtin_amdgcn_global_load_async_to_lds_b128)
#define HAVE_ASYNC_LDS 1
#endif
#if __has_builtin(__builtin_amdgcn_s_wait_asynccnt)
#define HAVE_WAIT_ASYNC 1
#endif
#endif

// builtin signature (from hipcc diagnostic): (v4i __device__*, v4i __shared__*, int, imm int)
typedef __attribute__((address_space(1))) v4i* gv4i_p;
typedef __attribute__((address_space(3))) v4i* lv4i_p;

__device__ inline unsigned short f2bf(float f) {
  // round-to-nearest-even f32 -> bf16 bits
  unsigned int u = __float_as_uint(f);
  u += 0x7fffu + ((u >> 16) & 1u);
  return (unsigned short)(u >> 16);
}

__device__ inline float gelu_exact(float x) {
  return 0.5f * x * (1.0f + erff(x * 0.70710678118654752440f));
}

// ---------------------------------------------------------------------------
// Prep: W1 (f32 [k][n]) -> W1T bf16 [n][k]   (B-operand: K contiguous per row)
//       W2 (f32 [k][8]) -> W2T bf16 [16][k]  (padded experts 8..15 = 0)
// ---------------------------------------------------------------------------
__global__ void prep_weights(const float* __restrict__ W1,
                             const float* __restrict__ W2,
                             unsigned short* __restrict__ W1T,
                             unsigned short* __restrict__ W2T) {
  int id = blockIdx.x * blockDim.x + threadIdx.x;
  if (id < D_MODEL * D_MODEL) {
    int n = id & (D_MODEL - 1);
    int k = id >> 10;                       // id = k*1024 + n (coalesced read)
    W1T[n * D_MODEL + k] = f2bf(W1[id]);
  } else {
    int j = id - D_MODEL * D_MODEL;
    if (j < 16 * D_MODEL) {
      int n = j >> 10;                      // padded expert row
      int k = j & (D_MODEL - 1);
      W2T[j] = (n < N_EXPERTS) ? f2bf(W2[k * N_EXPERTS + n]) : (unsigned short)0;
    }
  }
}

// ---------------------------------------------------------------------------
// Fused gate kernel: one workgroup = 32 tokens, 256 threads = 8 wave32.
// ---------------------------------------------------------------------------
__global__ __launch_bounds__(256) void moe_gate_kernel(
    const float* __restrict__ x,        // [N, D]
    const float* __restrict__ gamma,    // [D]
    const float* __restrict__ beta,     // [D]
    const float* __restrict__ b1,       // [D]
    const float* __restrict__ b2,       // [8]
    const unsigned short* __restrict__ W1T,  // bf16 [D][D], n-major
    const unsigned short* __restrict__ W2T,  // bf16 [16][D]
    float* __restrict__ out_routed,     // [N, 8]
    float* __restrict__ out_idx)        // [N, 2] (indices as float)
{
  extern __shared__ char smem[];
  unsigned short* xs  = (unsigned short*)smem;            // [32][1032] normalized x (bf16)
  unsigned short* w1s = xs + M_TILE * XS_STRIDE;          // [256][40]  W1 panel (n-major)
  unsigned short* hs  = w1s + NCHUNK * W1S_STRIDE;        // [32][264]  h chunk (bf16)

  const int tid  = threadIdx.x;
  const int wave = tid >> 5;
  const int lane = tid & 31;
  const int tok0 = blockIdx.x * M_TILE;

  // ----- Phase 1: LayerNorm (two-pass, wave-parallel) -> bf16 in LDS -----
  for (int t = 0; t < M_TILE / 8; ++t) {
    const int tok = wave * (M_TILE / 8) + t;
    const float* row = x + (size_t)(tok0 + tok) * D_MODEL;
    float v[32];
    float s = 0.f;
#pragma unroll
    for (int i = 0; i < 32; ++i) {       // lane-coalesced b32 loads
      v[i] = row[lane + 32 * i];
      s += v[i];
    }
#pragma unroll
    for (int off = 16; off > 0; off >>= 1) s += __shfl_xor(s, off, 32);
    const float mu = s * (1.0f / D_MODEL);
    float ss = 0.f;
#pragma unroll
    for (int i = 0; i < 32; ++i) { float d = v[i] - mu; ss += d * d; }
#pragma unroll
    for (int off = 16; off > 0; off >>= 1) ss += __shfl_xor(ss, off, 32);
    const float rs = rsqrtf(ss * (1.0f / D_MODEL) + LN_EPS);
#pragma unroll
    for (int i = 0; i < 32; ++i) {
      const int k = lane + 32 * i;
      xs[tok * XS_STRIDE + k] = f2bf((v[i] - mu) * rs * gamma[k] + beta[k]);
    }
  }
  __syncthreads();

  // Fragment coords per ISA 16-bit A/B VGPR layout:
  //   lanes 0-15 : row = lane,    K in {0..7} then {16..23}
  //   lanes 16-31: row = lane-16, K in {8..15} then {24..31}
  const int mrow  = lane & 15;
  const int kb_lo = (lane < 16) ? 0 : 8;

  // GEMM1 task: wave -> (m-tile, 4 n-tiles); A fragment reused 4x
  const int miA  = wave >> 2;          // 0..1
  const int niA0 = (wave & 3) * 4;     // n-tiles niA0 .. niA0+3
  // GEMM2 task: wave -> (m-tile, 2 K-steps per chunk)
  const int miB = wave & 1;
  const int kkB = wave >> 1;           // base K-step, also does kkB+4

  v8f logits_acc = {0.f,0.f,0.f,0.f,0.f,0.f,0.f,0.f};

  // panel staging: thread t stages one 32-k row (64 B) of the [256][32] panel
  const int sn = tid;                  // panel n-row 0..255

  for (int cn = 0; cn < D_MODEL / NCHUNK; ++cn) {
    const int n0 = cn * NCHUNK;
    v8f acc[4];
#pragma unroll
    for (int j = 0; j < 4; ++j) acc[j] = (v8f){0.f,0.f,0.f,0.f,0.f,0.f,0.f,0.f};

    for (int ks = 0; ks < D_MODEL / KSTEP; ++ks) {
      const int k0 = ks * KSTEP;
      // ---- stage W1T panel [NCHUNK][KSTEP] (16 KB, L2-resident source) ----
      const unsigned short* gsrc = W1T + (size_t)(n0 + sn) * D_MODEL + k0;
      unsigned short*       ldst = w1s + sn * W1S_STRIDE;
#ifdef HAVE_ASYNC_LDS
      // async direct global->LDS copy (ASYNCcnt-tracked); imm offset applies
      // to both the memory and LDS addresses (contiguous 64 B per thread)
      __builtin_amdgcn_global_load_async_to_lds_b128((gv4i_p)gsrc, (lv4i_p)ldst, 0, 0);
      __builtin_amdgcn_global_load_async_to_lds_b128((gv4i_p)gsrc, (lv4i_p)ldst, 16, 0);
      __builtin_amdgcn_global_load_async_to_lds_b128((gv4i_p)gsrc, (lv4i_p)ldst, 32, 0);
      __builtin_amdgcn_global_load_async_to_lds_b128((gv4i_p)gsrc, (lv4i_p)ldst, 48, 0);
#ifdef HAVE_WAIT_ASYNC
      __builtin_amdgcn_s_wait_asynccnt(0);
#else
      asm volatile("s_wait_asynccnt 0x0" ::: "memory");
#endif
#else
      uint4 q0 = *(const uint4*)(gsrc);
      uint4 q1 = *(const uint4*)(gsrc + 8);
      uint4 q2 = *(const uint4*)(gsrc + 16);
      uint4 q3 = *(const uint4*)(gsrc + 24);
      *(uint4*)(ldst)      = q0;
      *(uint4*)(ldst + 8)  = q1;
      *(uint4*)(ldst + 16) = q2;
      *(uint4*)(ldst + 24) = q3;
#endif
      if (ks + 1 < D_MODEL / KSTEP)
        __builtin_prefetch(gsrc + KSTEP, 0, 3);   // near-cache prefetch, next panel
      __syncthreads();

      // ---- fragments (2x b128 each; padded strides -> conflict-free) ----
      v16bf afrag;
      {
        const unsigned short* ap = xs + (miA * 16 + mrow) * XS_STRIDE + k0 + kb_lo;
        *((uint4*)&afrag)     = *(const uint4*)(ap);
        *((uint4*)&afrag + 1) = *(const uint4*)(ap + 16);
      }
      v16bf bfrag[4];
#pragma unroll
      for (int j = 0; j < 4; ++j) {
        const unsigned short* bp = w1s + ((niA0 + j) * 16 + mrow) * W1S_STRIDE + kb_lo;
        *((uint4*)&bfrag[j])     = *(const uint4*)(bp);
        *((uint4*)&bfrag[j] + 1) = *(const uint4*)(bp + 16);
      }
#pragma unroll
      for (int j = 0; j < 4; ++j)
        acc[j] = __builtin_amdgcn_wmma_f32_16x16x32_bf16(false, afrag, false, bfrag[j],
                                                         (short)0, acc[j], false, false);
      __syncthreads();
    }

    // ---- epilogue: h = GELU(acc + b1) -> bf16 chunk in LDS ----
    // C layout: VGPR r, lanes 0-15 -> (M=r, N=lane); lanes 16-31 -> (M=8+r, N=lane-16)
#pragma unroll
    for (int j = 0; j < 4; ++j) {
      const int ncol = (niA0 + j) * 16 + mrow;
      const float bv = b1[n0 + ncol];
      const int rbase = miA * 16 + ((lane < 16) ? 0 : 8);
#pragma unroll
      for (int r = 0; r < 8; ++r)
        hs[(rbase + r) * HS_STRIDE + ncol] = f2bf(gelu_exact(acc[j][r] + bv));
    }
    __syncthreads();

    // ---- GEMM2 partial: logits += h[:, chunk] @ W2T (2 K-slices per wave) ----
#pragma unroll
    for (int t = 0; t < 2; ++t) {
      const int kk = kkB + 4 * t;
      v16bf afrag, bfrag;
      const unsigned short* ap = hs + (miB * 16 + mrow) * HS_STRIDE + kk * KSTEP + kb_lo;
      *((uint4*)&afrag)     = *(const uint4*)(ap);
      *((uint4*)&afrag + 1) = *(const uint4*)(ap + 16);
      const unsigned short* bp = W2T + (size_t)mrow * D_MODEL + n0 + kk * KSTEP + kb_lo;
      *((uint4*)&bfrag)     = *(const uint4*)(bp);
      *((uint4*)&bfrag + 1) = *(const uint4*)(bp + 16);
      logits_acc = __builtin_amdgcn_wmma_f32_16x16x32_bf16(false, afrag, false, bfrag,
                                                           (short)0, logits_acc, false, false);
    }
    __syncthreads();   // hs reused next chunk
  }

  // ----- reduce per-wave partial logits, then softmax + top-2 -----
  float* lred = (float*)hs;   // 8 KB: [wave][vgpr r][lane]
#pragma unroll
  for (int r = 0; r < 8; ++r)
    lred[wave * 256 + r * 32 + lane] = logits_acc[r];
  __syncthreads();

  if (tid < M_TILE) {
    const int mi = tid >> 4;
    const int m  = tid & 15;
    const int rsel = m & 7;
    const int lb   = (m < 8) ? 0 : 16;
    float p[N_EXPERTS];
    float mx = -1e30f;
#pragma unroll
    for (int e = 0; e < N_EXPERTS; ++e) {
      float l = b2[e];
#pragma unroll
      for (int s = 0; s < 4; ++s)                // waves with (w&1)==mi
        l += lred[(mi + 2 * s) * 256 + rsel * 32 + lb + e];
      p[e] = l;
      mx = fmaxf(mx, l);
    }
    float sum = 0.f;
#pragma unroll
    for (int e = 0; e < N_EXPERTS; ++e) { p[e] = expf(p[e] - mx); sum += p[e]; }
    const float inv = 1.0f / sum;
#pragma unroll
    for (int e = 0; e < N_EXPERTS; ++e) p[e] *= inv;

    int i1 = 0;                                   // stable top-2 over 8
#pragma unroll
    for (int e = 1; e < N_EXPERTS; ++e) if (p[e] > p[i1]) i1 = e;
    int i2 = (i1 == 0) ? 1 : 0;
#pragma unroll
    for (int e = 0; e < N_EXPERTS; ++e) if (e != i1 && p[e] > p[i2]) i2 = e;

    const float denom = 1.0f / (p[i1] + p[i2] + RENORM_EPS);
    const size_t tg = (size_t)tok0 + tid;
#pragma unroll
    for (int e = 0; e < N_EXPERTS; ++e)
      out_routed[tg * N_EXPERTS + e] = (e == i1 || e == i2) ? p[e] * denom : 0.f;
    out_idx[tg * 2 + 0] = (float)i1;
    out_idx[tg * 2 + 1] = (float)i2;
  }
}

// ---------------------------------------------------------------------------
extern "C" void kernel_launch(void* const* d_in, const int* in_sizes, int n_in,
                              void* d_out, int out_size, void* d_ws, size_t ws_size,
                              hipStream_t stream) {
  (void)in_sizes; (void)n_in; (void)out_size; (void)ws_size;
  const float* x     = (const float*)d_in[0];
  const float* gamma = (const float*)d_in[1];
  const float* beta  = (const float*)d_in[2];
  const float* W1    = (const float*)d_in[3];
  const float* b1    = (const float*)d_in[4];
  const float* W2    = (const float*)d_in[5];
  const float* b2    = (const float*)d_in[6];

  // workspace: W1T bf16 (2 MB) + W2T bf16 padded (32 KB)
  unsigned short* W1T = (unsigned short*)d_ws;
  unsigned short* W2T = W1T + (size_t)D_MODEL * D_MODEL;

  const int prep_total = D_MODEL * D_MODEL + 16 * D_MODEL;
  prep_weights<<<(prep_total + 255) / 256, 256, 0, stream>>>(W1, W2, W1T, W2T);

  float* out_routed = (float*)d_out;
  float* out_idx    = out_routed + (size_t)N_TOKENS * N_EXPERTS;

  const size_t smem = (size_t)(M_TILE * XS_STRIDE + NCHUNK * W1S_STRIDE +
                               M_TILE * HS_STRIDE) * sizeof(unsigned short); // 103424 B
  moe_gate_kernel<<<N_TOKENS / M_TILE, 256, smem, stream>>>(
      x, gamma, beta, b1, b2, W1T, W2T, out_routed, out_idx);
}
